// MultiHeadSelfAttention_3298534883474
// MI455X (gfx1250) — matrix operand order repaired and checked by reading, not compiled, not run on hardware
//
#include <hip/hip_runtime.h>

typedef __attribute__((ext_vector_type(8)))  float   v8f;
typedef __attribute__((ext_vector_type(8)))  __bf16  v8bf;
typedef __attribute__((ext_vector_type(16))) __bf16  v16bf;
typedef uint4 __attribute__((may_alias)) uint4a;

#define EMBED 1024
#define HID   1024
#define NHEAD 16
#define BATCH 2
#define SEQ   2048
#define DHEAD 64
#define MROWS (BATCH*SEQ)   // 4096

union BfPun { uint4 u; v8bf b; };

__device__ __forceinline__ unsigned short f2bf(float f) {
  unsigned u = __float_as_uint(f);
  u += 0x7fffu + ((u >> 16) & 1u);   // round to nearest even
  return (unsigned short)(u >> 16);
}

__device__ __forceinline__ v16bf cat16(uint4 lo, uint4 hi) {
  BfPun a, b; a.u = lo; b.u = hi;
  return __builtin_shufflevector(a.b, b.b, 0,1,2,3,4,5,6,7,8,9,10,11,12,13,14,15);
}

__device__ __forceinline__ void storeOut(unsigned short* p, float v) { *p = f2bf(v); }
__device__ __forceinline__ void storeOut(float* p, float v) { *p = v; }

// --- CDNA5 async global->LDS path (ASYNCcnt-tracked, CDNA5 ISA) ------
// LDS destination passed as a REAL pointer: (a) generic-pointer low 32 bits
// are the LDS address (ISA 10.2 aperture rules), (b) the ptrtoint captures
// the __shared__ object so the "memory" clobber legally covers it and the
// subsequent ds_load fragment reads cannot be dead-code eliminated.
__device__ __forceinline__ void async_ld_b128(void* ldsDst, const void* g) {
  unsigned off = (unsigned)(size_t)ldsDst;
  asm volatile("global_load_async_to_lds_b128 %0, %1, off"
               :: "v"(off), "v"((unsigned long long)(size_t)g) : "memory");
}
__device__ __forceinline__ void async_ld_b64(void* ldsDst, const void* g) {
  unsigned off = (unsigned)(size_t)ldsDst;
  asm volatile("global_load_async_to_lds_b64 %0, %1, off"
               :: "v"(off), "v"((unsigned long long)(size_t)g) : "memory");
}
__device__ __forceinline__ void wait_async0() {
  asm volatile("s_wait_asynccnt 0" ::: "memory");
}

// ---------------------------------------------------------------------------
// f32 -> bf16 elementwise convert (X)
// ---------------------------------------------------------------------------
__global__ void cvt_x_kernel(const float* __restrict__ x,
                             unsigned short* __restrict__ xb, int nElem) {
  int i = (blockIdx.x * blockDim.x + threadIdx.x) * 4;
  if (i >= nElem) return;
  float4 v = *(const float4*)(x + i);
  ushort4 o;
  o.x = f2bf(v.x); o.y = f2bf(v.y); o.z = f2bf(v.z); o.w = f2bf(v.w);
  *(ushort4*)(xb + i) = o;
}

// ---------------------------------------------------------------------------
// Transpose (in,out)->(out,in) + convert f32 -> bf16 for the 4 weight matrices
// ---------------------------------------------------------------------------
__global__ void cvt_wT_kernel(const float* W0, const float* W1,
                              const float* W2, const float* W3,
                              unsigned short* T0, unsigned short* T1,
                              unsigned short* T2, unsigned short* T3) {
  const float* W; unsigned short* T;
  switch (blockIdx.z) {
    case 0:  W = W0; T = T0; break;
    case 1:  W = W1; T = T1; break;
    case 2:  W = W2; T = T2; break;
    default: W = W3; T = T3; break;
  }
  __shared__ float tile[32][33];
  int o0 = blockIdx.x * 32, i0 = blockIdx.y * 32;
  int tx = threadIdx.x, ty0 = threadIdx.y;      // blockDim = (32, 8)
#pragma unroll
  for (int k = 0; k < 4; ++k) {
    int ty = ty0 + k * 8;
    tile[ty][tx] = W[(size_t)(i0 + ty) * HID + (o0 + tx)];
  }
  __syncthreads();
#pragma unroll
  for (int k = 0; k < 4; ++k) {
    int ty = ty0 + k * 8;
    T[(size_t)(o0 + ty) * EMBED + (i0 + tx)] = f2bf(tile[tx][ty]);
  }
}

// ---------------------------------------------------------------------------
// bf16 WMMA GEMM: C = (A @ B + bias) * scale
//   A  : M x K row-major bf16
//   Bt : N x K row-major bf16 (B pre-transposed -> contiguous B fragments)
//   C  : M x N row-major (bf16 bits or f32)
// Block: 256 threads (8 waves), tile 128(M) x 256(N), K-step 32.
// A tile double-buffered in LDS, filled via async global->LDS (ASYNCcnt).
// Wave w owns N columns [w*32, w*32+32): 8(M) x 2(N) = 16 accumulators.
// ---------------------------------------------------------------------------
template <typename OutT>
__global__ void __launch_bounds__(256)
gemm_bf16_kernel(const unsigned short* __restrict__ A,
                 const unsigned short* __restrict__ Bt,
                 const float* __restrict__ bias,
                 OutT* __restrict__ C,
                 float scale, int M, int N, int K) {
  // 112B row stride: 16B-aligned, conflict-free b128 reads
  __shared__ __align__(16) unsigned short sA[2][128][56];
  const int tid  = threadIdx.x;
  const int wave = tid >> 5, lane = tid & 31;
  const int hh = lane >> 4, nn = lane & 15;
  const int m0 = blockIdx.x * 128;
  const int n0 = blockIdx.y * 256;

  auto stageA = [&](int k0, int buf) {
    for (int c = tid; c < 512; c += 256) {             // 16B chunks
      int row = c >> 2, ko = (c & 3) << 3;
      async_ld_b128(&sA[buf][row][ko], &A[(size_t)(m0 + row) * K + k0 + ko]);
    }
  };

  const v8f vzero = {0.f,0.f,0.f,0.f,0.f,0.f,0.f,0.f};
  v8f acc[8][2];
#pragma unroll
  for (int i = 0; i < 8; ++i) { acc[i][0] = vzero; acc[i][1] = vzero; }

  stageA(0, 0);                                        // prologue fill
  const int nk = K >> 5;
  for (int ks = 0; ks < nk; ++ks) {
    const int k0 = ks << 5, cur = ks & 1;
    wait_async0();                                     // buf[cur] resident
    __syncthreads();
    if (ks + 1 < nk) stageA(k0 + 32, cur ^ 1);         // overlap next fill

    // B fragments direct from global (lane: col = nn, K = k0 + 16*hh + 0..15)
    v16bf bfr[2];
#pragma unroll
    for (int s = 0; s < 2; ++s) {
      const unsigned short* bp =
          Bt + (size_t)(n0 + wave * 32 + s * 16 + nn) * K + k0 + hh * 16;
      bfr[s] = cat16(*(const uint4a*)bp, *(const uint4a*)(bp + 8));
      if (k0 + 32 < K) __builtin_prefetch(bp + 32, 0, 0);  // global_prefetch_b8
    }

#pragma unroll
    for (int mt = 0; mt < 8; ++mt) {
      int row = mt * 16 + nn;
      // A layout: elems 0..15 -> K = 16h+0..15
      v16bf af = cat16(*(const uint4a*)&sA[cur][row][hh * 16],
                       *(const uint4a*)&sA[cur][row][hh * 16 + 8]);
      acc[mt][0] = __builtin_amdgcn_wmma_f32_16x16x32_bf16(
          false, af, false, bfr[0], (short)0, acc[mt][0], false, false);
      acc[mt][1] = __builtin_amdgcn_wmma_f32_16x16x32_bf16(
          false, af, false, bfr[1], (short)0, acc[mt][1], false, false);
    }
    __syncthreads();                                   // done reading buf[cur]
  }

  // epilogue: C/D layout -> lane nn = N, row = mt*16 + j + 8*hh
#pragma unroll
  for (int mt = 0; mt < 8; ++mt) {
#pragma unroll
    for (int s = 0; s < 2; ++s) {
      int col = n0 + wave * 32 + s * 16 + nn;
      float bv = bias[col];
#pragma unroll
      for (int j = 0; j < 8; ++j) {
        int row = m0 + mt * 16 + j + hh * 8;
        storeOut(&C[(size_t)row * N + col], (acc[mt][s][j] + bv) * scale);
      }
    }
  }
}

// ---------------------------------------------------------------------------
// Attention diagonal kernel.
// Per head (n,e): Qr/Kr/Vr are the CONTIGUOUS 2048x64 slabs at head*128*1024
// (raw reshape => no gather). S = Qr @ Kr^T; softmax over the query axis per
// column b; O[b,:] = w_b * V[b,:] with w_b = exp(S[b,b]-max)/sum.
// Block = 256 threads, 128 b-columns (16 per wave). K-fragments are
// loop-invariant registers; Q tiles double-buffered via async global->LDS.
// ---------------------------------------------------------------------------
__global__ void __launch_bounds__(256)
attn_diag_kernel(const unsigned short* __restrict__ Qb,
                 const unsigned short* __restrict__ Kb,
                 const unsigned short* __restrict__ Vb,
                 unsigned short* __restrict__ Ob) {
  __shared__ __align__(16) unsigned short sQ[2][16][72];  // 144B stride
  const int tid  = threadIdx.x;
  const int wave = tid >> 5, lane = tid & 31;
  const int hh = lane >> 4, nn = lane & 15;
  const size_t base = (size_t)blockIdx.y * 128 * 1024;    // head slab (elems)
  const int b0   = blockIdx.x * 128 + wave * 16;
  const int bcol = b0 + nn;

  auto stageQ = [&](int a0, int buf) {                    // 8B per thread
    int row = tid >> 4, ko = (tid & 15) << 2;
    async_ld_b64(&sQ[buf][row][ko], &Qb[base + (size_t)(a0 + row) * 64 + ko]);
  };

  // Loop-invariant B operand: Kr rows for our 16 columns ([N][K] layout natively)
  const unsigned short* kp = Kb + base + (size_t)bcol * 64;
  const v16bf bk0 = cat16(*(const uint4a*)(kp + hh * 16),
                          *(const uint4a*)(kp + hh * 16 + 8));
  const v16bf bk1 = cat16(*(const uint4a*)(kp + 32 + hh * 16),
                          *(const uint4a*)(kp + 32 + hh * 16 + 8));
  const v8f vzero = {0.f,0.f,0.f,0.f,0.f,0.f,0.f,0.f};

  float m = -1e30f, ssum = 0.f, diag = 0.f;

  stageQ(0, 0);                                           // prologue fill
  for (int step = 0; step < SEQ / 16; ++step) {
    const int a0 = step << 4, cur = step & 1;
    wait_async0();                                        // buf[cur] resident
    __syncthreads();
    if (step + 1 < SEQ / 16) stageQ(a0 + 16, cur ^ 1);    // overlap next fill

    v16bf qa = cat16(*(const uint4a*)&sQ[cur][nn][hh * 16],
                     *(const uint4a*)&sQ[cur][nn][hh * 16 + 8]);
    v16bf qb = cat16(*(const uint4a*)&sQ[cur][nn][32 + hh * 16],
                     *(const uint4a*)&sQ[cur][nn][32 + hh * 16 + 8]);
    v8f st = __builtin_amdgcn_wmma_f32_16x16x32_bf16(
        false, qa, false, bk0, (short)0, vzero, false, false);
    st = __builtin_amdgcn_wmma_f32_16x16x32_bf16(
        false, qb, false, bk1, (short)0, st, false, false);

    // online softmax stats over the row (query) axis for our column
    float tmax = -1e30f;
#pragma unroll
    for (int j = 0; j < 8; ++j) tmax = fmaxf(tmax, st[j]);
    tmax = fmaxf(tmax, __shfl_xor(tmax, 16, 32));
    float nm = fmaxf(m, tmax);
    float p = 0.f;
#pragma unroll
    for (int j = 0; j < 8; ++j) p += __expf(st[j] - nm);
    p += __shfl_xor(p, 16, 32);
    ssum = ssum * __expf(m - nm) + p;
    m = nm;

    if (a0 == b0) {   // diagonal element lives in this tile: row j+8*hh == nn
      float d = 0.f;
#pragma unroll
      for (int j = 0; j < 8; ++j) d += (j + hh * 8 == nn) ? st[j] : 0.f;
      d += __shfl_xor(d, 16, 32);
      diag = d;
    }
    __syncthreads();                                      // done reading buf[cur]
  }

  const float w = __expf(diag - m) / ssum;

  // O[b, :] = w * V[b, :]  (lane pair splits the 64-wide row)
  const unsigned short* vp = Vb + base + (size_t)bcol * 64 + hh * 32;
  unsigned short* op = Ob + base + (size_t)bcol * 64 + hh * 32;
#pragma unroll
  for (int c = 0; c < 4; ++c) {
    BfPun in, out; in.u = *(const uint4a*)(vp + c * 8);
#pragma unroll
    for (int j = 0; j < 8; ++j) out.b[j] = (__bf16)((float)in.b[j] * w);
    *(uint4a*)(op + c * 8) = out.u;
  }
}

// ---------------------------------------------------------------------------
extern "C" void kernel_launch(void* const* d_in, const int* in_sizes, int n_in,
                              void* d_out, int out_size, void* d_ws, size_t ws_size,
                              hipStream_t stream) {
  const float* X  = (const float*)d_in[0];
  const float* Wq = (const float*)d_in[1];
  const float* bq = (const float*)d_in[2];
  const float* Wk = (const float*)d_in[3];
  const float* bk = (const float*)d_in[4];
  const float* Wv = (const float*)d_in[5];
  const float* bv = (const float*)d_in[6];
  const float* Wo = (const float*)d_in[7];
  const float* bo = (const float*)d_in[8];
  float* out = (float*)d_out;

  // workspace layout (bf16 = unsigned short), total 48 MiB
  unsigned short* Xb  = (unsigned short*)d_ws;            // 4096x1024
  unsigned short* Wqt = Xb  + (size_t)MROWS * EMBED;      // 1024x1024 each
  unsigned short* Wkt = Wqt + (size_t)EMBED * HID;
  unsigned short* Wvt = Wkt + (size_t)EMBED * HID;
  unsigned short* Wot = Wvt + (size_t)EMBED * HID;
  unsigned short* Qb  = Wot + (size_t)EMBED * HID;        // 4096x1024 each
  unsigned short* Kb  = Qb  + (size_t)MROWS * HID;
  unsigned short* Vb  = Kb  + (size_t)MROWS * HID;
  unsigned short* Ob  = Vb  + (size_t)MROWS * HID;

  const float qk_scale = 1.0f / 32.0f;    // 1/sqrt(H), applied to both Q and K

  // 1) convert inputs to bf16 (weights also transposed to [out][in])
  cvt_x_kernel<<<(MROWS * EMBED) / (256 * 4), 256, 0, stream>>>(X, Xb, MROWS * EMBED);
  cvt_wT_kernel<<<dim3(HID / 32, EMBED / 32, 4), dim3(32, 8), 0, stream>>>(
      Wq, Wk, Wv, Wo, Wqt, Wkt, Wvt, Wot);

  // 2) Q/K/V projections (bias fused, 1/sqrt(H) folded into Q and K)
  dim3 ggrid(MROWS / 128, HID / 256);
  gemm_bf16_kernel<unsigned short><<<ggrid, 256, 0, stream>>>(
      Xb, Wqt, bq, Qb, qk_scale, MROWS, HID, EMBED);
  gemm_bf16_kernel<unsigned short><<<ggrid, 256, 0, stream>>>(
      Xb, Wkt, bk, Kb, qk_scale, MROWS, HID, EMBED);
  gemm_bf16_kernel<unsigned short><<<ggrid, 256, 0, stream>>>(
      Xb, Wvt, bv, Vb, 1.0f, MROWS, HID, EMBED);

  // 3) per-head column-softmax diagonal weights + O = w * V
  attn_diag_kernel<<<dim3(SEQ / 128, BATCH * NHEAD), 256, 0, stream>>>(Qb, Kb, Vb, Ob);

  // 4) output projection -> f32 d_out
  gemm_bf16_kernel<float><<<ggrid, 256, 0, stream>>>(
      Ob, Wot, bo, out, 1.0f, MROWS, HID, HID);
}
